// GraphConvNetLSTM_81183471829489
// MI455X (gfx1250) — matrix operand order repaired and checked by reading, not compiled, not run on hardware
//
#include <hip/hip_runtime.h>
#include <hip/hip_bf16.h>

// ---------------------------------------------------------------------------
// Types for CDNA5 WMMA
// ---------------------------------------------------------------------------
typedef __attribute__((ext_vector_type(16))) __bf16 v16bf;
typedef __attribute__((ext_vector_type(8)))  float  v8f;

#define N_NODES 4096
#define FEAT    1024
#define NPAD    16      // GCN width 10 padded to 16 (cols 10..15 kept at 0)

__device__ __forceinline__ unsigned short f2bf_u16(float f) {
    __bf16 b = (__bf16)f;                       // RNE; lets clang emit v_cvt_*bf16*
    return __builtin_bit_cast(unsigned short, b);
}
__device__ __forceinline__ unsigned int pack2bf(float lo, float hi) {
    return (unsigned int)f2bf_u16(lo) | ((unsigned int)f2bf_u16(hi) << 16);
}
__device__ __forceinline__ float seluf(float v) {
    const float kScale = 1.0507009873554805f;
    const float kAlpha = 1.6732632423543772f;
    return v > 0.0f ? kScale * v : kScale * kAlpha * (__expf(v) - 1.0f);
}
__device__ __forceinline__ float sigf(float z) {
    return 1.0f / (1.0f + __expf(-z));
}

// ---------------------------------------------------------------------------
// 1) s[r] = deg_r > 0 ? 1/sqrt(deg_r) : 0,  deg = row sums of A
// ---------------------------------------------------------------------------
__global__ __launch_bounds__(256) void k_rowsum(const float* __restrict__ A,
                                                float* __restrict__ s) {
    __shared__ float red[256];
    const int r = blockIdx.x;
    float acc = 0.0f;
    const float* row = A + (size_t)r * N_NODES;
    for (int k = threadIdx.x; k < N_NODES; k += 256) acc += row[k];
    red[threadIdx.x] = acc;
    __syncthreads();
    for (int off = 128; off > 0; off >>= 1) {
        if (threadIdx.x < off) red[threadIdx.x] += red[threadIdx.x + off];
        __syncthreads();
    }
    if (threadIdx.x == 0) {
        float d = red[0];
        s[r] = (d > 0.0f) ? rsqrtf(d) : 0.0f;
    }
}

// ---------------------------------------------------------------------------
// 2) One-time A f32 -> bf16 (row-major, same layout); lives in L2 (32 MB)
// ---------------------------------------------------------------------------
__global__ __launch_bounds__(256) void k_convA(const float* __restrict__ A,
                                               unsigned short* __restrict__ Abf) {
    const size_t i = ((size_t)blockIdx.x * 256 + threadIdx.x) * 8;
    const float4* ap = (const float4*)(A + i);
    float4 a0 = ap[0], a1 = ap[1];
    uint4 o;
    o.x = pack2bf(a0.x, a0.y); o.y = pack2bf(a0.z, a0.w);
    o.z = pack2bf(a1.x, a1.y); o.w = pack2bf(a1.z, a1.w);
    *(uint4*)(Abf + i) = o;
}

// ---------------------------------------------------------------------------
// 3) G0 = X @ W_gcn0^T   (4096x1024)x(1024x10) -> (4096x16), cols>=10 zero
// ---------------------------------------------------------------------------
__global__ __launch_bounds__(256) void k_xw0(const float* __restrict__ X,
                                             const float* __restrict__ W0,
                                             float* __restrict__ G) {
    const int tid = threadIdx.x;
    const int c = tid & 15;
    const int r = blockIdx.x * 16 + (tid >> 4);
    float acc = 0.0f;
    if (c < 10) {
        const float4* xr = (const float4*)(X + (size_t)r * FEAT);
        const float4* wr = (const float4*)(W0 + (size_t)c * FEAT);
        for (int k = 0; k < FEAT / 4; ++k) {
            float4 xv = xr[k], wv = wr[k];
            acc += xv.x * wv.x + xv.y * wv.y + xv.z * wv.z + xv.w * wv.w;
        }
    }
    G[(size_t)r * NPAD + c] = acc;
}

// ---------------------------------------------------------------------------
// 4) pack:  G[r,c] = (W ? sum_j Hin[r,j]*W[c,j] : Hin[r,c])   (W is 10x10)
//           Bt[c][r] = bf16( s[r] * G[r,c] )   (transposed, WMMA-B friendly)
// ---------------------------------------------------------------------------
__global__ __launch_bounds__(256) void k_pack(const float* __restrict__ Hin,
                                              const float* __restrict__ W,
                                              const float* __restrict__ s,
                                              float* __restrict__ Gout,
                                              unsigned short* __restrict__ Bt) {
    const int tid = threadIdx.x;
    const int c = tid & 15;
    const int r = blockIdx.x * 16 + (tid >> 4);
    float g = 0.0f;
    if (W) {
        if (c < 10) {
            const float* hr = Hin + (size_t)r * NPAD;
            const float* wr = W + c * 10;
            #pragma unroll
            for (int j = 0; j < 10; ++j) g += hr[j] * wr[j];
        }
    } else {
        g = Hin[(size_t)r * NPAD + c];
    }
    Gout[(size_t)r * NPAD + c] = g;
    Bt[(size_t)c * N_NODES + r] = f2bf_u16(s[r] * g);
}

// ---------------------------------------------------------------------------
// 5a) WMMA GEMM, pre-converted bf16 A (fast path):
//     P = Abf @ Bt^T ;  Hout = selu(s_r*P + s_r^2*G)
//     One 16x16 M-tile per wave; K=4096 in 128 steps of wmma_f32_16x16x32_bf16
// ---------------------------------------------------------------------------
__global__ __launch_bounds__(256) void k_gemm_bf16(const unsigned short* __restrict__ Abf,
                                                   const unsigned short* __restrict__ Bt,
                                                   const float* __restrict__ G,
                                                   const float* __restrict__ s,
                                                   float* __restrict__ Hout) {
    const int wave = threadIdx.x >> 5;           // 0..7
    const int lane = threadIdx.x & 31;
    const int tile = blockIdx.x * 8 + wave;      // 0..255
    const int row0 = tile * 16;
    const int hf   = lane >> 4;                  // lane half (0/1)
    const int ln   = lane & 15;
    const unsigned short* arow = Abf + (size_t)(row0 + ln) * N_NODES;
    const unsigned short* brow = Bt + (size_t)ln * N_NODES;

    v8f acc = {};
    #pragma unroll 4
    for (int k0 = 0; k0 < N_NODES; k0 += 32) {
        __builtin_prefetch((const void*)(arow + k0 + 1024), 0, 1);
        // A fragment (16x32 bf16): VGPR0-3 = K phase hf*8..+7, VGPR4-7 = +16
        union { uint4 q[2]; v16bf v; } af;
        af.q[0] = *(const uint4*)(arow + k0 + hf * 8);
        af.q[1] = *(const uint4*)(arow + k0 + 16 + hf * 8);
        // B fragment (32x16 bf16): lane ln = column N, K phase hf*8..+7, then +16
        union { uint4 q[2]; v16bf v; } bfu;
        bfu.q[0] = *(const uint4*)(brow + k0 + hf * 8);
        bfu.q[1] = *(const uint4*)(brow + k0 + 16 + hf * 8);
        acc = __builtin_amdgcn_wmma_f32_16x16x32_bf16(
            false, af.v, false, bfu.v, (short)0, acc, false, false);
    }

    #pragma unroll
    for (int v = 0; v < 8; ++v) {
        const int row = row0 + hf * 8 + v;
        const int col = ln;
        const float sv = s[row];
        const float val = sv * acc[v] + sv * sv * G[(size_t)row * NPAD + col];
        Hout[(size_t)row * NPAD + col] = seluf(val);
    }
}

// ---------------------------------------------------------------------------
// 5b) WMMA GEMM, on-the-fly f32->bf16 A conversion (workspace-lean fallback)
// ---------------------------------------------------------------------------
__global__ __launch_bounds__(256) void k_gemm_f32(const float* __restrict__ A,
                                                  const unsigned short* __restrict__ Bt,
                                                  const float* __restrict__ G,
                                                  const float* __restrict__ s,
                                                  float* __restrict__ Hout) {
    const int wave = threadIdx.x >> 5;
    const int lane = threadIdx.x & 31;
    const int tile = blockIdx.x * 8 + wave;
    const int row0 = tile * 16;
    const int hf   = lane >> 4;
    const int ln   = lane & 15;
    const float* arow = A + (size_t)(row0 + ln) * N_NODES;
    const unsigned short* brow = Bt + (size_t)ln * N_NODES;

    v8f acc = {};
    #pragma unroll 2
    for (int k0 = 0; k0 < N_NODES; k0 += 32) {
        const float4* ap0 = (const float4*)(arow + k0 + hf * 8);
        const float4* ap1 = (const float4*)(arow + k0 + 16 + hf * 8);
        float4 a0 = ap0[0], a1 = ap0[1];
        float4 a2 = ap1[0], a3 = ap1[1];
        union { unsigned int u[8]; v16bf v; } af;
        af.u[0] = pack2bf(a0.x, a0.y); af.u[1] = pack2bf(a0.z, a0.w);
        af.u[2] = pack2bf(a1.x, a1.y); af.u[3] = pack2bf(a1.z, a1.w);
        af.u[4] = pack2bf(a2.x, a2.y); af.u[5] = pack2bf(a2.z, a2.w);
        af.u[6] = pack2bf(a3.x, a3.y); af.u[7] = pack2bf(a3.z, a3.w);
        union { uint4 q[2]; v16bf v; } bfu;
        bfu.q[0] = *(const uint4*)(brow + k0 + hf * 8);
        bfu.q[1] = *(const uint4*)(brow + k0 + 16 + hf * 8);
        acc = __builtin_amdgcn_wmma_f32_16x16x32_bf16(
            false, af.v, false, bfu.v, (short)0, acc, false, false);
    }

    #pragma unroll
    for (int v = 0; v < 8; ++v) {
        const int row = row0 + hf * 8 + v;
        const int col = ln;
        const float sv = s[row];
        const float val = sv * acc[v] + sv * sv * G[(size_t)row * NPAD + col];
        Hout[(size_t)row * NPAD + col] = seluf(val);
    }
}

// ---------------------------------------------------------------------------
// 6) Sequential 3-layer bidirectional LSTM: 4096 encode + 4096 decode steps.
//    Single wave32 workgroup; weights resident in LDS; barriers are S_NOP.
// ---------------------------------------------------------------------------
__global__ __launch_bounds__(32) void k_lstm(const float* __restrict__ H,
                                             const float* __restrict__ Wih,
                                             const float* __restrict__ Whh,
                                             const float* __restrict__ bih,
                                             const float* __restrict__ bhh,
                                             const float* __restrict__ h0,
                                             const float* __restrict__ c0,
                                             const float* __restrict__ tok,
                                             const float* __restrict__ Wo,
                                             const float* __restrict__ bo,
                                             float* __restrict__ probs) {
    __shared__ float sWih[6][20][10];
    __shared__ float sWhh[6][20][5];
    __shared__ float sB[6][20];
    __shared__ float h[6][5], c[6][5];
    __shared__ float x[10];
    __shared__ float g[2][20];
    const int tid = threadIdx.x;

    for (int i = tid; i < 1200; i += 32) ((float*)sWih)[i] = Wih[i];
    for (int i = tid; i <  600; i += 32) ((float*)sWhh)[i] = Whh[i];
    for (int i = tid; i <  120; i += 32) ((float*)sB)[i]   = bih[i] + bhh[i];
    for (int i = tid; i <   30; i += 32) { ((float*)h)[i] = h0[i]; ((float*)c)[i] = c0[i]; }
    __syncthreads();

    for (int t = 0; t < 2 * N_NODES; ++t) {
        if (t < N_NODES) {
            if (tid < 10) x[tid] = H[(size_t)t * NPAD + tid];     // encode input
        } else if (t == N_NODES) {
            if (tid < 10) x[tid] = tok[tid];                      // decode seed
        }
        __syncthreads();
        #pragma unroll
        for (int l = 0; l < 3; ++l) {
            for (int idx = tid; idx < 40; idx += 32) {            // 40 gate rows
                const int d = idx / 20, row = idx % 20, k = 2 * l + d;
                float acc = sB[k][row];
                #pragma unroll
                for (int j = 0; j < 10; ++j) acc += sWih[k][row][j] * x[j];
                #pragma unroll
                for (int j = 0; j < 5; ++j)  acc += sWhh[k][row][j] * h[k][j];
                g[d][row] = acc;
            }
            __syncthreads();
            if (tid < 10) {                                       // cell update
                const int d = tid / 5, j = tid % 5, k = 2 * l + d;
                const float ig = sigf(g[d][j]);
                const float fg = sigf(g[d][5 + j]);
                const float gg = tanhf(g[d][10 + j]);
                const float og = sigf(g[d][15 + j]);
                const float cn = fg * c[k][j] + ig * gg;
                c[k][j] = cn;
                const float hn = og * tanhf(cn);
                h[k][j] = hn;
                x[d * 5 + j] = hn;                                // feedback
            }
            __syncthreads();
        }
        if (t >= N_NODES && tid == 0) {                           // output head
            float acc = bo[0];
            #pragma unroll
            for (int i = 0; i < 10; ++i) acc += x[i] * Wo[i];
            probs[t - N_NODES] = sigf(acc);
        }
    }
}

// ---------------------------------------------------------------------------
// Host side
// ---------------------------------------------------------------------------
extern "C" void kernel_launch(void* const* d_in, const int* in_sizes, int n_in,
                              void* d_out, int out_size, void* d_ws, size_t ws_size,
                              hipStream_t stream) {
    const float* A    = (const float*)d_in[0];
    const float* X    = (const float*)d_in[1];
    const float* W0   = (const float*)d_in[2];
    const float* W1   = (const float*)d_in[3];
    const float* W2   = (const float*)d_in[4];
    const float* Wih  = (const float*)d_in[5];
    const float* Whh  = (const float*)d_in[6];
    const float* bih  = (const float*)d_in[7];
    const float* bhh  = (const float*)d_in[8];
    const float* h0   = (const float*)d_in[9];
    const float* c0   = (const float*)d_in[10];
    const float* tok  = (const float*)d_in[11];
    const float* Wo   = (const float*)d_in[12];
    const float* bo   = (const float*)d_in[13];
    float* out = (float*)d_out;

    char* ws = (char*)d_ws;
    float*          s    = (float*)ws;                              //  16 KB
    float*          bufA = (float*)(ws + (16 << 10));               // 256 KB
    float*          bufB = (float*)(ws + (16 << 10) + (256 << 10)); // 256 KB
    unsigned short* Bt   = (unsigned short*)(ws + (16 << 10) + (512 << 10)); // 128 KB
    unsigned short* Abf  = (unsigned short*)(ws + (1 << 20));       //  32 MB (optional)

    const size_t abf_bytes = (size_t)N_NODES * N_NODES * 2;
    const bool use_bf16A = ws_size >= ((size_t)(1 << 20) + abf_bytes);

    // Normalization vector s = deg^{-1/2}
    k_rowsum<<<N_NODES, 256, 0, stream>>>(A, s);

    if (use_bf16A) {
        // One-time A conversion; thereafter A traffic is bf16 out of L2.
        k_convA<<<(N_NODES * (N_NODES / 8)) / 256, 256, 0, stream>>>(A, Abf);
    }

    // Layer 0: G0 = X @ W0^T
    k_xw0<<<N_NODES / 16, 256, 0, stream>>>(X, W0, bufA);
    k_pack<<<N_NODES / 16, 256, 0, stream>>>(bufA, nullptr, s, bufA, Bt);
    if (use_bf16A) k_gemm_bf16<<<32, 256, 0, stream>>>(Abf, Bt, bufA, s, bufB);
    else           k_gemm_f32 <<<32, 256, 0, stream>>>(A,   Bt, bufA, s, bufB);

    // Layer 1
    k_pack<<<N_NODES / 16, 256, 0, stream>>>(bufB, W1, s, bufA, Bt);
    if (use_bf16A) k_gemm_bf16<<<32, 256, 0, stream>>>(Abf, Bt, bufA, s, bufB);
    else           k_gemm_f32 <<<32, 256, 0, stream>>>(A,   Bt, bufA, s, bufB);

    // Layer 2
    k_pack<<<N_NODES / 16, 256, 0, stream>>>(bufB, W2, s, bufA, Bt);
    if (use_bf16A) k_gemm_bf16<<<32, 256, 0, stream>>>(Abf, Bt, bufA, s, bufB);
    else           k_gemm_f32 <<<32, 256, 0, stream>>>(A,   Bt, bufA, s, bufB);

    // Encode + autoregressive decode; writes probs to d_out
    k_lstm<<<1, 32, 0, stream>>>(bufB, Wih, Whh, bih, bhh, h0, c0, tok, Wo, bo, out);
}
